// MultiHeadAttention_28432683500192
// MI455X (gfx1250) — compile-verified, matrix operand order checked by reading, not run on hardware
//
#include <hip/hip_runtime.h>
#include <hip/hip_bf16.h>

typedef __attribute__((ext_vector_type(16))) _Float16 v16h;
typedef __attribute__((ext_vector_type(8)))  float    v8f;
typedef __attribute__((ext_vector_type(4)))  _Float16 h4;
typedef __attribute__((ext_vector_type(4)))  unsigned int u32x4;
typedef __attribute__((ext_vector_type(8)))  int i32x8;
typedef __attribute__((ext_vector_type(4)))  int i32x4;

#define BB  4
#define LL  2048
#define DD  1024
#define HH  16
#define DKK 64
#define BL  (BB * LL)

// log2(e) / sqrt(dk): softmax computed in exp2 domain (v_exp_f32 is native exp2)
#define SCLOG2E 0.1803368801111f
#define NEGBIG  (-3.0e38f)

// ---------------------------------------------------------------------------
// 16x32 f16 WMMA fragment load (A layout; B identical when source is stored
// column-major so the lane index selects a column).
// ---------------------------------------------------------------------------
__device__ __forceinline__ v16h load_frag_h16(const _Float16* base, int stride,
                                              int row, int kb, int koff) {
  v16h f;
  const _Float16* p = base + row * stride + koff + kb;
#pragma unroll
  for (int v = 0; v < 4; ++v) {
    f[2 * v]     = p[2 * v];
    f[2 * v + 1] = p[2 * v + 1];
    f[8 + 2 * v] = p[16 + 2 * v];
    f[9 + 2 * v] = p[16 + 2 * v + 1];
  }
  return f;
}

__device__ __forceinline__ unsigned int lds_offset_of(const void* p) {
  // generic (flat) pointers to LDS carry the byte offset in the low 32 bits
  return (unsigned int)(uintptr_t)p;
}

// ---------------------------------------------------------------------------
// Tensor Data Mover: 2-D f16 tile  global -> LDS  with per-row LDS padding.
//   tile_x  elements per row (contiguous), tile_y rows
//   srow    global stride between rows (elements)
//   pad_iv  pad_interval code (3 = every 16 dwords, 4 = every 32 dwords)
// pad_amount code 0 = 1 dword of padding  ->  row stride += 4 bytes in LDS.
// Descriptor per CDNA5 D# spec (group0 128b, group1 256b, groups 2/3 zeroed:
// 2-D tile). workgroup_mask = 0 (not dispatched in a cluster).
// ---------------------------------------------------------------------------
__device__ __forceinline__ void tdm_load_2d_f16(const _Float16* gsrc,
                                                unsigned int lds_off,
                                                unsigned int tile_x, unsigned int tile_y,
                                                unsigned int tensor_x, unsigned int tensor_y,
                                                unsigned long long srow,
                                                unsigned int pad_iv) {
  unsigned long long ga = (unsigned long long)(uintptr_t)gsrc;
  u32x4 g0;
  g0[0] = 1u;                                              // count=1
  g0[1] = lds_off;                                         // lds_addr
  g0[2] = (unsigned int)ga;                                // global_addr[31:0]
  g0[3] = (unsigned int)((ga >> 32) & 0x1FFFFFFu) | (2u << 30);  // [56:32] | type=2

  i32x8 g1;
  g1[0] = (int)((1u << 16) |                               // data_size = 2 bytes
                (1u << 20) |                               // pad_enable
                (pad_iv << 22));                           // pad_interval; pad_amount=0 (1 dw)
  g1[1] = (int)((tensor_x & 0xFFFFu) << 16);               // tensor_dim0[15:0]
  g1[2] = (int)(((tensor_x >> 16) & 0xFFFFu) | ((tensor_y & 0xFFFFu) << 16));
  g1[3] = (int)(((tensor_y >> 16) & 0xFFFFu) | ((tile_x & 0xFFFFu) << 16));
  g1[4] = (int)(tile_y & 0xFFFFu);                         // tile_dim1; tile_dim2=0
  g1[5] = (int)(unsigned int)(srow & 0xFFFFFFFFull);       // tensor_dim0_stride[31:0]
  g1[6] = (int)(unsigned int)((srow >> 32) & 0xFFFFull);   // [47:32]; dim1_stride[15:0]=0
  g1[7] = 0;

  i32x4 gz = {0, 0, 0, 0};
#if __clang_major__ >= 23
  i32x8 gz8 = {0, 0, 0, 0, 0, 0, 0, 0};
  __builtin_amdgcn_tensor_load_to_lds(g0, g1, gz, gz, gz8, 0);
#else
  __builtin_amdgcn_tensor_load_to_lds(g0, g1, gz, gz, 0);
#endif
}

// ---------------------------------------------------------------------------
// Tiled GEMM:  C(M,N) = A(M,K) * W(N,K)^T + bias   (torch Linear semantics)
// 256 threads = 8 waves; tile 128(M) x 64(N); K stepped by 32; each wave
// computes one 16x64 strip via v_wmma_f32_16x16x32_f16.
// MODE 0: fp32 row-major out | 1: f16 (B,H,L,DK) | 2: f16 (B,H,DK,L)
// AF16:   A source is f16 workspace -> staged by the Tensor Data Mover.
// ---------------------------------------------------------------------------
template <int MODE, bool AF16>
__global__ __launch_bounds__(256) void gemm_wmma(const void* __restrict__ Aptr,
                                                 const float* __restrict__ W,
                                                 const float* __restrict__ bias,
                                                 void* __restrict__ Cptr) {
  constexpr int K = DD;
  __shared__ _Float16 Alds[128][34];  // 17-dword row stride: conflict-free
  __shared__ _Float16 Wlds[64][34];

  const int tid   = threadIdx.x;
  const int wave  = tid >> 5;
  const int lane  = tid & 31;
  const int row16 = lane & 15;
  const int kb    = (lane >> 4) << 3;
  const int tileM = blockIdx.y * 128;
  const int tileN = blockIdx.x * 64;

  v8f acc[4] = {};

  for (int k0 = 0; k0 < K; k0 += 32) {
    // ---- stage A tile (128x32) ----
    if constexpr (AF16) {
      // TDM: each wave DMAs its own 16-row slice (TENSOR ops ignore EXEC, so
      // per-wave slicing avoids duplicate issue). pad code 3 = pad after every
      // 16 dwords (32 f16) -> 34-half LDS row stride.
      const _Float16* Ag = (const _Float16*)Aptr;
      tdm_load_2d_f16(Ag + (size_t)(tileM + wave * 16) * K + k0,
                      lds_offset_of(&Alds[wave * 16][0]),
                      /*tile*/ 32, 16, /*tensor*/ 32, 16, /*srow*/ (unsigned long long)K,
                      /*pad_iv*/ 3);
    } else {
      const float* Ag = (const float*)Aptr;
#pragma unroll
      for (int i = 0; i < 4; ++i) {
        int idx = tid + i * 256;
        int r = idx >> 3, c = (idx & 7) << 2;
        float4 d = *(const float4*)(Ag + (size_t)(tileM + r) * K + k0 + c);
        Alds[r][c]     = (_Float16)d.x;
        Alds[r][c + 1] = (_Float16)d.y;
        Alds[r][c + 2] = (_Float16)d.z;
        Alds[r][c + 3] = (_Float16)d.w;
      }
    }
    // ---- stage W tile (64x32), fp32 -> f16 on VALU ----
#pragma unroll
    for (int i = 0; i < 2; ++i) {
      int idx = tid + i * 256;
      int r = idx >> 3, c = (idx & 7) << 2;
      float4 d = *(const float4*)(W + (size_t)(tileN + r) * K + k0 + c);
      Wlds[r][c]     = (_Float16)d.x;
      Wlds[r][c + 1] = (_Float16)d.y;
      Wlds[r][c + 2] = (_Float16)d.z;
      Wlds[r][c + 3] = (_Float16)d.w;
    }
    // ---- prefetch next K-step into L2 ----
    if (k0 + 32 < K) {
      if constexpr (!AF16) {
        const float* Ag = (const float*)Aptr;
        __builtin_prefetch(Ag + (size_t)(tileM + (tid >> 1)) * K + k0 + 32 + (tid & 1) * 16, 0, 0);
      }
      if (tid < 128)
        __builtin_prefetch(W + (size_t)(tileN + (tid >> 1)) * K + k0 + 32 + (tid & 1) * 16, 0, 0);
    }
    if constexpr (AF16) __builtin_amdgcn_s_wait_tensorcnt(0);
    __syncthreads();

    // ---- one A frag, 4 B frags, 4 WMMAs ----
    v16h afrag = load_frag_h16(&Alds[0][0], 34, wave * 16 + row16, kb, 0);
#pragma unroll
    for (int n = 0; n < 4; ++n) {
      v16h bfrag = load_frag_h16(&Wlds[0][0], 34, n * 16 + row16, kb, 0);
      acc[n] = __builtin_amdgcn_wmma_f32_16x16x32_f16(false, afrag, false, bfrag,
                                                      (short)0, acc[n], false, false);
    }
    __syncthreads();
  }

  // ---- epilogue: bias + layout-specific store ----
  const int mbase = tileM + wave * 16 + ((lane >> 4) << 3);
#pragma unroll
  for (int n = 0; n < 4; ++n) {
    const int col = tileN + n * 16 + row16;
    const float bv = bias[col];
#pragma unroll
    for (int v = 0; v < 8; ++v) {
      const int row = mbase + v;
      const float val = acc[n][v] + bv;
      if constexpr (MODE == 0) {
        ((float*)Cptr)[(size_t)row * DD + col] = val;
      } else if constexpr (MODE == 1) {
        const int b = row >> 11, l = row & (LL - 1);
        const int h = col >> 6,  d = col & 63;
        ((_Float16*)Cptr)[(((size_t)(b * HH + h) * LL + l) << 6) + d] = (_Float16)val;
      } else {
        const int b = row >> 11, l = row & (LL - 1);
        const int h = col >> 6,  d = col & 63;
        ((_Float16*)Cptr)[((size_t)(b * HH + h) * DKK + d) * LL + l] = (_Float16)val;
      }
    }
  }
}

// ---------------------------------------------------------------------------
// Flash attention: one block = 4 waves = 64 queries of one (b,h).
// K/V tiles staged by the Tensor Data Mover (one 16-row slice per wave),
// S = Q*K^T via WMMA, online softmax in exp2 domain, P transposed through
// per-wave LDS, ctx += P*V via WMMA. Scores never touch HBM.
// ---------------------------------------------------------------------------
__global__ __launch_bounds__(128) void attn_flash(const _Float16* __restrict__ Qh,
                                                  const _Float16* __restrict__ Kh,
                                                  const _Float16* __restrict__ Vt,
                                                  const int* __restrict__ mask,
                                                  _Float16* __restrict__ Ccat) {
  __shared__ _Float16 Klds[64][66];       // [key][dk], 33-dword row stride
  __shared__ _Float16 Vlds[64][66];       // [dk][key]
  __shared__ _Float16 Plds[4][16][66];    // per-wave P transpose buffer

  const int tid   = threadIdx.x;
  const int wave  = tid >> 5;
  const int lane  = tid & 31;
  const int row16 = lane & 15;
  const int kb    = (lane >> 4) << 3;
  const int bh    = blockIdx.y;
  const int b     = bh >> 4;
  const int h     = bh & 15;
  const int qtile = blockIdx.x * 64;

  // Q fragments (16 queries x 64 dk) pinned in registers for the whole pass
  const _Float16* Qbase = Qh + ((size_t)bh * LL + qtile + wave * 16) * DKK;
  const v16h aq0 = load_frag_h16(Qbase, DKK, row16, kb, 0);
  const v16h aq1 = load_frag_h16(Qbase, DKK, row16, kb, 32);

  v8f ctx[4] = {};
  float mrun[8], lrun[8];
#pragma unroll
  for (int v = 0; v < 8; ++v) { mrun[v] = NEGBIG; lrun[v] = 0.f; }

  const _Float16* Kb = Kh + (size_t)bh * LL * DKK;
  const _Float16* Vb = Vt + (size_t)bh * DKK * LL;

  for (int kt = 0; kt < LL; kt += 64) {
    // ---- TDM-stage K (key-major) and V (dk-major): 16 rows per wave.
    // pad code 4 = pad after every 32 dwords (64 f16) -> 66-half row stride.
    tdm_load_2d_f16(Kb + (size_t)(kt + wave * 16) * DKK,
                    lds_offset_of(&Klds[wave * 16][0]),
                    64, 16, 64, 16, (unsigned long long)DKK, 4);
    tdm_load_2d_f16(Vb + (size_t)(wave * 16) * LL + kt,
                    lds_offset_of(&Vlds[wave * 16][0]),
                    64, 16, 64, 16, (unsigned long long)LL, 4);
    if (kt + 64 < LL) {
      __builtin_prefetch(Kb + (size_t)(kt + 64 + (tid >> 1)) * DKK + (tid & 1) * 32, 0, 0);
      __builtin_prefetch(Vb + (size_t)(tid >> 1) * LL + kt + 64 + (tid & 1) * 32, 0, 0);
    }
    __builtin_amdgcn_s_wait_tensorcnt(0);
    __syncthreads();

    // ---- S = (Q*K^T) in exp2 domain, key-padding mask ----
    v8f s[4];
#pragma unroll
    for (int n = 0; n < 4; ++n) {
      v16h bk0 = load_frag_h16(&Klds[0][0], 66, n * 16 + row16, kb, 0);
      v16h bk1 = load_frag_h16(&Klds[0][0], 66, n * 16 + row16, kb, 32);
      v8f z = {};
      z = __builtin_amdgcn_wmma_f32_16x16x32_f16(false, aq0, false, bk0, (short)0, z, false, false);
      z = __builtin_amdgcn_wmma_f32_16x16x32_f16(false, aq1, false, bk1, (short)0, z, false, false);
      const int kidx = kt + n * 16 + row16;          // this lane's key column
      const bool live = mask[b * LL + kidx] != 0;
#pragma unroll
      for (int v = 0; v < 8; ++v) s[n][v] = live ? z[v] * SCLOG2E : NEGBIG;
    }

    // ---- online softmax: rows live in lane-halves; xor<16 stays in-half ----
    float mx[8];
#pragma unroll
    for (int v = 0; v < 8; ++v)
      mx[v] = fmaxf(fmaxf(s[0][v], s[1][v]), fmaxf(s[2][v], s[3][v]));
#pragma unroll
    for (int off = 1; off < 16; off <<= 1)
#pragma unroll
      for (int v = 0; v < 8; ++v)
        mx[v] = fmaxf(mx[v], __shfl_xor(mx[v], off, 32));

    float rs[8];
#pragma unroll
    for (int v = 0; v < 8; ++v) {
      const float mnew = fmaxf(mrun[v], mx[v]);
      const float sc = __builtin_amdgcn_exp2f(mrun[v] - mnew);
      mrun[v] = mnew;
      lrun[v] *= sc;
      rs[v] = 0.f;
#pragma unroll
      for (int n = 0; n < 4; ++n) {
        const float p = __builtin_amdgcn_exp2f(s[n][v] - mnew);
        s[n][v] = p;
        rs[v] += p;
      }
#pragma unroll
      for (int n = 0; n < 4; ++n) ctx[n][v] *= sc;
    }
#pragma unroll
    for (int off = 1; off < 16; off <<= 1)
#pragma unroll
      for (int v = 0; v < 8; ++v)
        rs[v] += __shfl_xor(rs[v], off, 32);
#pragma unroll
    for (int v = 0; v < 8; ++v) lrun[v] += rs[v];

    // ---- transpose P from C layout to A layout via per-wave LDS ----
    const int mtop = (lane >> 4) << 3;
#pragma unroll
    for (int n = 0; n < 4; ++n)
#pragma unroll
      for (int v = 0; v < 8; ++v)
        Plds[wave][mtop + v][n * 16 + row16] = (_Float16)s[n][v];
    // same-wave DS RAW is kept in order by hardware (DScnt in-order)

    v16h ap0 = load_frag_h16(&Plds[wave][0][0], 66, row16, kb, 0);
    v16h ap1 = load_frag_h16(&Plds[wave][0][0], 66, row16, kb, 32);

    // ---- ctx += P * V ----
#pragma unroll
    for (int n = 0; n < 4; ++n) {
      v16h bv0 = load_frag_h16(&Vlds[0][0], 66, n * 16 + row16, kb, 0);
      v16h bv1 = load_frag_h16(&Vlds[0][0], 66, n * 16 + row16, kb, 32);
      ctx[n] = __builtin_amdgcn_wmma_f32_16x16x32_f16(false, ap0, false, bv0, (short)0, ctx[n], false, false);
      ctx[n] = __builtin_amdgcn_wmma_f32_16x16x32_f16(false, ap1, false, bv1, (short)0, ctx[n], false, false);
    }
    __syncthreads();
  }

  // ---- normalize and store into concat (B,L,D) layout as f16 ----
  const int mtop = (lane >> 4) << 3;
#pragma unroll
  for (int v = 0; v < 8; ++v) {
    const int q = qtile + wave * 16 + mtop + v;
    const float inv = 1.0f / fmaxf(lrun[v], 1e-30f);
    _Float16* orow = Ccat + ((size_t)b * LL + q) * DD + h * DKK;
#pragma unroll
    for (int n = 0; n < 4; ++n)
      orow[n * 16 + row16] = (_Float16)(ctx[n][v] * inv);
  }
}

// ---------------------------------------------------------------------------
// Launcher. Workspace layout (f16): Qh | Kh | Vt | concat, 16 MB each (64 MB).
// ---------------------------------------------------------------------------
extern "C" void kernel_launch(void* const* d_in, const int* in_sizes, int n_in,
                              void* d_out, int out_size, void* d_ws, size_t ws_size,
                              hipStream_t stream) {
  (void)in_sizes; (void)n_in; (void)out_size; (void)ws_size;
  const float* q   = (const float*)d_in[0];
  const float* k   = (const float*)d_in[1];
  const float* v   = (const float*)d_in[2];
  const float* w_q = (const float*)d_in[3];
  const float* b_q = (const float*)d_in[4];
  const float* w_k = (const float*)d_in[5];
  const float* b_k = (const float*)d_in[6];
  const float* w_v = (const float*)d_in[7];
  const float* b_v = (const float*)d_in[8];
  const float* w_o = (const float*)d_in[9];
  const float* b_o = (const float*)d_in[10];
  const int*  mask = (const int*)d_in[11];
  float* out = (float*)d_out;

  _Float16* wsh = (_Float16*)d_ws;
  const size_t HSZ = (size_t)BB * HH * LL * DKK;   // 8,388,608 elems = 16 MB f16
  _Float16* Qh = wsh;
  _Float16* Kh = wsh + HSZ;
  _Float16* Vt = wsh + 2 * HSZ;
  _Float16* Cc = wsh + 3 * HSZ;

  dim3 gblock(256);
  dim3 ggrid(DD / 64, BL / 128);   // 16 x 64 blocks

  gemm_wmma<1, false><<<ggrid, gblock, 0, stream>>>(q, w_q, b_q, Qh);
  gemm_wmma<1, false><<<ggrid, gblock, 0, stream>>>(k, w_k, b_k, Kh);
  gemm_wmma<2, false><<<ggrid, gblock, 0, stream>>>(v, w_v, b_v, Vt);
  attn_flash<<<dim3(LL / 64, BB * HH), dim3(128), 0, stream>>>(Qh, Kh, Vt, mask, Cc);
  gemm_wmma<0, true><<<ggrid, gblock, 0, stream>>>(Cc, w_o, b_o, out);
}
